// Model_65317862637912
// MI455X (gfx1250) — matrix operand order repaired and checked by reading, not compiled, not run on hardware
//
#include <hip/hip_runtime.h>
#include <math.h>

// ---------------- constants ----------------
#define BB   8
#define L0   1024
#define CIN  64
#define DM   512
#define NH   8
#define DH   64
#define DFF  2048

typedef __attribute__((ext_vector_type(16))) __bf16 v16bf;
typedef __attribute__((ext_vector_type(8)))  __bf16 v8bf;
typedef __attribute__((ext_vector_type(8)))  float  v8f;

#define HAVE_ASYNC_LDS __has_builtin(__builtin_amdgcn_global_load_async_to_lds_b128)

// ---------------- helpers ----------------
static __device__ __forceinline__ float warp_sum(float v) {
  v += __shfl_xor(v, 16, 32);
  v += __shfl_xor(v, 8, 32);
  v += __shfl_xor(v, 4, 32);
  v += __shfl_xor(v, 2, 32);
  v += __shfl_xor(v, 1, 32);
  return v;
}
static __device__ __forceinline__ float warp_max(float v) {
  v = fmaxf(v, __shfl_xor(v, 16, 32));
  v = fmaxf(v, __shfl_xor(v, 8, 32));
  v = fmaxf(v, __shfl_xor(v, 4, 32));
  v = fmaxf(v, __shfl_xor(v, 2, 32));
  v = fmaxf(v, __shfl_xor(v, 1, 32));
  return v;
}
static __device__ __forceinline__ unsigned hashu(unsigned x) {
  x ^= x >> 16; x *= 0x7feb352dU;
  x ^= x >> 15; x *= 0x846ca68bU;
  x ^= x >> 16;
  return x;
}

#if HAVE_ASYNC_LDS
// b128 payload type matching the builtin's expected parameter:
// pointer to 16-byte int vector, global (AS1) source, LDS (AS3) destination.
typedef int b128_t __attribute__((vector_size(16)));
typedef __attribute__((address_space(1))) b128_t* gptr128_t;
typedef __attribute__((address_space(3))) b128_t* lptr128_t;

static __device__ __forceinline__ void async_copy16(__bf16* lds, const __bf16* g) {
  __builtin_amdgcn_global_load_async_to_lds_b128(
      (gptr128_t)(g), (lptr128_t)(lds), 0, 0);
}
static __device__ __forceinline__ void wait_async_le(int n) {
#if __has_builtin(__builtin_amdgcn_s_wait_asynccnt)
  if (n == 0) __builtin_amdgcn_s_wait_asynccnt(0);
  else        __builtin_amdgcn_s_wait_asynccnt(1);
#else
  if (n == 0) asm volatile("s_wait_asynccnt 0x0" ::: "memory");
  else        asm volatile("s_wait_asynccnt 0x1" ::: "memory");
#endif
}
#endif

// ---------------- f32 -> bf16 conversion ----------------
__global__ void k_cvt_bf16(const float* __restrict__ src, __bf16* __restrict__ dst, int n)
{
  int i = (blockIdx.x * blockDim.x + threadIdx.x) * 4;
  if (i >= n) return;
  float4 v = *(const float4*)(src + i);
  dst[i + 0] = (__bf16)v.x;
  dst[i + 1] = (__bf16)v.y;
  dst[i + 2] = (__bf16)v.z;
  dst[i + 3] = (__bf16)v.w;
}

// ---------------- WMMA GEMM (bf16 in, f32 out): C = act(A @ W^T + bias + resid) ----------------
// A [M,K] bf16 row-major, W [N,K] bf16 row-major.
// block = 256 threads = 8 waves; wave computes 16(M) x 64(N); block covers 128 x 64.
// B tile (64 x 32 bf16 per K-step) is staged in LDS: async double-buffered copy when the
// gfx1250 async-to-LDS builtin exists, ds_store fallback otherwise.
// Requires M % 128 == 0, N % 64 == 0, K % 32 == 0.
#define BSTRIDE 40   // 32 K elements + 8 pad -> 80B rows: 16B aligned, conflict-free b128 reads
__global__ __launch_bounds__(256)
void k_gemm(const __bf16* __restrict__ A, const __bf16* __restrict__ W,
            const float* __restrict__ bias, const float* __restrict__ resid,
            float* __restrict__ C, int M, int N, int K, int act)
{
  __shared__ __bf16 Bs[2][64 * BSTRIDE];

  const int lane = threadIdx.x & 31;
  const int wave = threadIdx.x >> 5;
  const int half = lane >> 4;
  const int l16  = lane & 15;
  const int mBase = blockIdx.y * 128 + wave * 16;
  const int nBase = blockIdx.x * 64;

  v8f acc[4];
#pragma unroll
  for (int j = 0; j < 4; ++j)
#pragma unroll
    for (int r = 0; r < 8; ++r) acc[j][r] = 0.f;

  // cooperative B-tile copy: thread t moves 16B -> row tn, K-segment tk
  const int tn = threadIdx.x >> 2;         // 0..63
  const int tk = (threadIdx.x & 3) * 8;    // 0,8,16,24
  const __bf16* wsrc = W + (size_t)(nBase + tn) * K + tk;

  const __bf16* arow = A + (size_t)(mBase + l16) * K;

#if HAVE_ASYNC_LDS
  async_copy16(&Bs[0][tn * BSTRIDE + tk], wsrc);        // prologue: tile 0
#endif

  for (int kk = 0; kk < K; kk += 32) {
#if HAVE_ASYNC_LDS
    const int buf = (kk >> 5) & 1;
    if (kk + 32 < K) {
      async_copy16(&Bs[buf ^ 1][tn * BSTRIDE + tk], wsrc + kk + 32);
      wait_async_le(1);                                 // current tile done
    } else {
      wait_async_le(0);
    }
    __syncthreads();                                    // whole tile visible to all waves
#else
    const int buf = 0;
    uint4 wv = *(const uint4*)(wsrc + kk);
    __syncthreads();                                    // protect previous reads
    *(uint4*)&Bs[0][tn * BSTRIDE + tk] = wv;
    __syncthreads();
#endif

    // A fragment: lane row M=l16; K {0..7,16..23} (half 0) / {8..15,24..31} (half 1)
    const __bf16* ap = arow + kk + half * 8;
    v8bf alo = *(const v8bf*)ap;
    v8bf ahi = *(const v8bf*)(ap + 16);
    v16bf af = __builtin_shufflevector(alo, ahi, 0, 1, 2, 3, 4, 5, 6, 7,
                                       8, 9, 10, 11, 12, 13, 14, 15);
    __builtin_prefetch(ap + 32, 0, 0);                  // global_prefetch_b8: next A tile

#pragma unroll
    for (int j = 0; j < 4; ++j) {
      // B fragment from LDS: lane col N=l16; K {0..7,16..23} (half 0) / {8..15,24..31} (half 1)
      const __bf16* bp = &Bs[buf][(j * 16 + l16) * BSTRIDE + half * 8];
      v8bf blo = *(const v8bf*)bp;
      v8bf bhi = *(const v8bf*)(bp + 16);
      v16bf bfv = __builtin_shufflevector(blo, bhi, 0, 1, 2, 3, 4, 5, 6, 7,
                                          8, 9, 10, 11, 12, 13, 14, 15);
      acc[j] = __builtin_amdgcn_wmma_f32_16x16x32_bf16(
          false, af, false, bfv, (short)0, acc[j], false, false);
    }
#if HAVE_ASYNC_LDS
    __syncthreads();    // all reads of `buf` done before it is overwritten next+1 iter
#endif
  }

  // epilogue: D layout lane = N col, VGPR r -> M row r + 8*half
#pragma unroll
  for (int j = 0; j < 4; ++j) {
    const int n = nBase + j * 16 + l16;
    const float bv = bias ? bias[n] : 0.f;
#pragma unroll
    for (int r = 0; r < 8; ++r) {
      const int m = mBase + r + half * 8;
      float v = acc[j][r] + bv;
      if (resid) v += resid[(size_t)m * N + n];
      if (act == 1) v = v > 0.f ? v : 0.f;
      C[(size_t)m * N + n] = v;
    }
  }
}

// ---------------- embedding: circular conv1d(64->512,k=3) + positional encoding ----------------
__global__ void k_embed(const float* __restrict__ x, const float* __restrict__ w,
                        float* __restrict__ X, int L)
{
  int idx = blockIdx.x * blockDim.x + threadIdx.x;     // over B*L*DM
  if (idx >= BB * L * DM) return;
  int n = idx % DM;
  int m = idx / DM;
  int l = m % L;
  int b = m / L;
  const float* wr = w + (size_t)n * (CIN * 3);
  float acc = 0.f;
#pragma unroll
  for (int t = 0; t < 3; ++t) {
    int ls = l + t - 1;
    ls = (ls < 0) ? ls + L : (ls >= L ? ls - L : ls);
    const float* xr = x + ((size_t)b * L + ls) * CIN;
    for (int c = 0; c < CIN; ++c)
      acc += (xr[c] + 1e-10f) * wr[c * 3 + t];
  }
  int j = n >> 1;
  float div = __expf(-(float)(2 * j) * (9.210340371976184f / 512.f)); // ln(1e4)/512
  float ang = (float)l * div;
  float pe = (n & 1) ? __cosf(ang) : __sinf(ang);
  X[idx] = acc + pe;
}

// ---------------- layernorm over last dim (512) ----------------
__global__ __launch_bounds__(256)
void k_layernorm(const float* __restrict__ A,
                 const float* __restrict__ g, const float* __restrict__ bta,
                 float* __restrict__ O)
{
  __shared__ float red[8];
  const int row = blockIdx.x;
  const int t = threadIdx.x;
  float v0 = A[(size_t)row * DM + t];
  float v1 = A[(size_t)row * DM + t + 256];

  float s = warp_sum(v0 + v1);
  if ((t & 31) == 0) red[t >> 5] = s;
  __syncthreads();
  float mean = 0.f;
#pragma unroll
  for (int i = 0; i < 8; ++i) mean += red[i];
  mean *= (1.f / 512.f);
  __syncthreads();

  float d0 = v0 - mean, d1 = v1 - mean;
  float q = warp_sum(d0 * d0 + d1 * d1);
  if ((t & 31) == 0) red[t >> 5] = q;
  __syncthreads();
  float var = 0.f;
#pragma unroll
  for (int i = 0; i < 8; ++i) var += red[i];
  var *= (1.f / 512.f);
  float rs = rsqrtf(var + 1e-5f);
  O[(size_t)row * DM + t]       = d0 * rs * g[t] + bta[t];
  O[(size_t)row * DM + t + 256] = d1 * rs * g[t + 256] + bta[t + 256];
}

// ---------------- ProbSparse: sampled sparsity measure M per (b,h,l) ----------------
__global__ void k_probe_scores(const float* __restrict__ Q, const float* __restrict__ Km,
                               float* __restrict__ Mm, int L, int U, int layer)
{
  int idx = blockIdx.x * blockDim.x + threadIdx.x;     // (b,h,l)
  if (idx >= BB * NH * L) return;
  int l = idx % L;
  int h = (idx / L) % NH;
  int b = idx / (L * NH);
  const float* qr = Q + ((size_t)b * L + l) * DM + h * DH;
  float q[DH];
#pragma unroll
  for (int d = 0; d < DH; ++d) q[d] = qr[d];
  float mx = -1e30f, sm = 0.f;
  for (int u = 0; u < U; ++u) {
    unsigned r = hashu(((unsigned)layer * 131071u) ^
                       ((unsigned)l * 2654435761u) ^
                       ((unsigned)u * 40503u + 0x9e3779b9u));
    int kidx = (int)(r % (unsigned)L);
    const float* kr = Km + ((size_t)b * L + kidx) * DM + h * DH;
    float s = 0.f;
#pragma unroll
    for (int d = 0; d < DH; ++d) s += q[d] * kr[d];
    mx = fmaxf(mx, s);
    sm += s;
  }
  Mm[idx] = mx - sm / (float)L;
}

// ---------------- top-U selection per (b,h) (iterative argmax; destroys Mm) ----------------
__global__ __launch_bounds__(256)
void k_topu(float* __restrict__ Mm, int* __restrict__ top, int L, int U)
{
  __shared__ float sv[256];
  __shared__ int   si[256];
  const int bh = blockIdx.x;
  float* mrow = Mm + (size_t)bh * L;
  for (int t = 0; t < U; ++t) {
    float best = -1e30f; int bi = 0;
    for (int l = threadIdx.x; l < L; l += 256) {
      float v = mrow[l];
      if (v > best) { best = v; bi = l; }
    }
    sv[threadIdx.x] = best; si[threadIdx.x] = bi;
    __syncthreads();
    for (int off = 128; off; off >>= 1) {
      if (threadIdx.x < off) {
        float ov = sv[threadIdx.x + off]; int oi = si[threadIdx.x + off];
        if (ov > sv[threadIdx.x] ||
            (ov == sv[threadIdx.x] && oi < si[threadIdx.x])) {
          sv[threadIdx.x] = ov; si[threadIdx.x] = oi;
        }
      }
      __syncthreads();
    }
    if (threadIdx.x == 0) { top[bh * 64 + t] = si[0]; mrow[si[0]] = -1e30f; }
    __syncthreads();
  }
}

// ---------------- scores for top-u queries vs all keys ----------------
__global__ void k_attn_scores(const float* __restrict__ Q, const float* __restrict__ Km,
                              const int* __restrict__ top, float* __restrict__ S,
                              int L, int U)
{
  int idx = blockIdx.x * blockDim.x + threadIdx.x;     // (bh,u,l)
  if (idx >= BB * NH * U * L) return;
  int l = idx % L;
  int u = (idx / L) % U;
  int bh = idx / (L * U);
  int b = bh / NH, h = bh % NH;
  int qi = top[bh * 64 + u];
  const float* qr = Q + ((size_t)b * L + qi) * DM + h * DH;
  const float* kr = Km + ((size_t)b * L + l) * DM + h * DH;
  float s = 0.f;
#pragma unroll
  for (int d = 0; d < DH; ++d) s += qr[d] * kr[d];
  S[((size_t)bh * U + u) * L + l] = s * 0.125f;        // 1/sqrt(64)
}

// ---------------- row softmax (row length L) ----------------
__global__ __launch_bounds__(256)
void k_softmax(float* __restrict__ S, int L)
{
  __shared__ float red[8];
  float* row = S + (size_t)blockIdx.x * L;
  int t = threadIdx.x;
  float mx = -1e30f;
  for (int l = t; l < L; l += 256) mx = fmaxf(mx, row[l]);
  mx = warp_max(mx);
  if ((t & 31) == 0) red[t >> 5] = mx;
  __syncthreads();
  float m = -1e30f;
#pragma unroll
  for (int i = 0; i < 8; ++i) m = fmaxf(m, red[i]);
  __syncthreads();
  float s = 0.f;
  for (int l = t; l < L; l += 256) { float e = __expf(row[l] - m); row[l] = e; s += e; }
  s = warp_sum(s);
  if ((t & 31) == 0) red[t >> 5] = s;
  __syncthreads();
  float tot = 0.f;
#pragma unroll
  for (int i = 0; i < 8; ++i) tot += red[i];
  float inv = 1.f / tot;
  for (int l = t; l < L; l += 256) row[l] *= inv;
}

// ---------------- mean of V over sequence, per (b, h*64+d) ----------------
__global__ void k_meanv(const float* __restrict__ V, float* __restrict__ meanV, int L)
{
  int idx = blockIdx.x * blockDim.x + threadIdx.x;     // b*DM + n
  if (idx >= BB * DM) return;
  int b = idx / DM, n = idx % DM;
  float s = 0.f;
  for (int l = 0; l < L; ++l) s += V[((size_t)b * L + l) * DM + n];
  meanV[idx] = s / (float)L;
}

__global__ void k_fill_ctx(const float* __restrict__ meanV, float* __restrict__ ctx, int L)
{
  int idx = blockIdx.x * blockDim.x + threadIdx.x;     // M*DM
  if (idx >= BB * L * DM) return;
  int n = idx % DM;
  int m = idx / DM;
  int b = m / L;
  ctx[idx] = meanV[b * DM + n];
}

// ---------------- scatter ctx[top[u]] = attn @ V ----------------
__global__ void k_ctx_update(const float* __restrict__ S, const float* __restrict__ V,
                             const int* __restrict__ top, float* __restrict__ ctx,
                             int L, int U)
{
  int idx = blockIdx.x * blockDim.x + threadIdx.x;     // (bh,u,d)
  if (idx >= BB * NH * U * DH) return;
  int d = idx % DH;
  int u = (idx / DH) % U;
  int bh = idx / (DH * U);
  int b = bh / NH, h = bh % NH;
  const float* srow = S + ((size_t)bh * U + u) * L;
  const float* vcol = V + (size_t)b * L * DM + h * DH + d;
  float s = 0.f;
  for (int l = 0; l < L; ++l) s += srow[l] * vcol[(size_t)l * DM];
  int qi = top[bh * 64 + u];
  ctx[((size_t)b * L + qi) * DM + h * DH + d] = s;
}

// ---------------- distil: circular gather into bf16 [M, 1536] (col = c*3 + t) ----------------
__global__ void k_gather3(const float* __restrict__ X, __bf16* __restrict__ Xc, int L)
{
  int idx = blockIdx.x * blockDim.x + threadIdx.x;     // M*1536
  if (idx >= BB * L * 1536) return;
  int col = idx % 1536;
  int m = idx / 1536;
  int c = col / 3, t = col % 3;
  int l = m % L, b = m / L;
  int ls = l + t - 1;
  ls = (ls < 0) ? ls + L : (ls >= L ? ls - L : ls);
  Xc[idx] = (__bf16)X[((size_t)b * L + ls) * DM + c];
}

// ---------------- batchnorm stats per channel ----------------
__global__ __launch_bounds__(256)
void k_bn_stats(const float* __restrict__ Y, float* __restrict__ mu,
                float* __restrict__ var, int Mrows)
{
  __shared__ float r1[8], r2[8];
  int n = blockIdx.x;
  float s = 0.f, q = 0.f;
  for (int m = threadIdx.x; m < Mrows; m += 256) {
    float v = Y[(size_t)m * DM + n];
    s += v; q += v * v;
  }
  s = warp_sum(s); q = warp_sum(q);
  int t = threadIdx.x;
  if ((t & 31) == 0) { r1[t >> 5] = s; r2[t >> 5] = q; }
  __syncthreads();
  if (t == 0) {
    float ts = 0.f, tq = 0.f;
#pragma unroll
    for (int i = 0; i < 8; ++i) { ts += r1[i]; tq += r2[i]; }
    float m_ = ts / (float)Mrows;
    mu[n] = m_;
    var[n] = tq / (float)Mrows - m_ * m_;
  }
}

// ---------------- BN apply + ELU + maxpool(3,2,pad 1) ----------------
__global__ void k_bn_elu_pool(const float* __restrict__ Y, const float* __restrict__ mu,
                              const float* __restrict__ var, const float* __restrict__ g,
                              const float* __restrict__ be, float* __restrict__ Xo, int L)
{
  int Lo = L >> 1;
  int idx = blockIdx.x * blockDim.x + threadIdx.x;     // BB*Lo*DM
  if (idx >= BB * Lo * DM) return;
  int n = idx % DM;
  int mo = idx / DM;
  int io = mo % Lo, b = mo / Lo;
  float m_ = mu[n], rs = rsqrtf(var[n] + 1e-5f), gg = g[n], bb = be[n];
  float best = -1e30f;
#pragma unroll
  for (int t = -1; t <= 1; ++t) {
    int j = 2 * io + t;
    if (j >= 0 && j < L) {
      float v = (Y[((size_t)b * L + j) * DM + n] - m_) * rs * gg + bb;
      v = v > 0.f ? v : (__expf(v) - 1.f);
      best = fmaxf(best, v);
    }
  }
  Xo[idx] = best;
}

// ---------------- final mean over sequence ----------------
__global__ void k_mean(const float* __restrict__ X, float* __restrict__ out, int L)
{
  int idx = blockIdx.x * blockDim.x + threadIdx.x;     // BB*DM
  if (idx >= BB * DM) return;
  int b = idx / DM, n = idx % DM;
  float s = 0.f;
  for (int l = 0; l < L; ++l) s += X[((size_t)b * L + l) * DM + n];
  out[idx] = s / (float)L;
}

// ---------------- host orchestration ----------------
static inline void cvt(const float* s, __bf16* d, size_t n, hipStream_t st) {
  k_cvt_bf16<<<(unsigned)((n / 4 + 255) / 256), 256, 0, st>>>(s, d, (int)n);
}

extern "C" void kernel_launch(void* const* d_in, const int* in_sizes, int n_in,
                              void* d_out, int out_size, void* d_ws, size_t ws_size,
                              hipStream_t stream)
{
  (void)in_sizes; (void)n_in; (void)out_size; (void)ws_size;
  const float* x_enc = (const float*)d_in[0];
  const float* tok_w = (const float*)d_in[1];
  const float* Wq = (const float*)d_in[2];
  const float* bq = (const float*)d_in[3];
  const float* Wk = (const float*)d_in[4];
  const float* bk = (const float*)d_in[5];
  const float* Wv = (const float*)d_in[6];
  const float* bv = (const float*)d_in[7];
  const float* Wo = (const float*)d_in[8];
  const float* bo = (const float*)d_in[9];
  const float* c1w = (const float*)d_in[10];
  const float* c1b = (const float*)d_in[11];
  const float* c2w = (const float*)d_in[12];
  const float* c2b = (const float*)d_in[13];
  const float* n1g = (const float*)d_in[14];
  const float* n1b = (const float*)d_in[15];
  const float* n2g = (const float*)d_in[16];
  const float* n2b = (const float*)d_in[17];
  const float* dcw = (const float*)d_in[18];
  const float* dcb = (const float*)d_in[19];
  const float* bng = (const float*)d_in[20];
  const float* bnb = (const float*)d_in[21];
  const float* fng = (const float*)d_in[22];
  const float* fnb = (const float*)d_in[23];
  float* out = (float*)d_out;

  char* ws = (char*)d_ws;
  size_t off = 0;
  auto allocf = [&](size_t nf) -> float* {
    float* p = (float*)(ws + off);
    off += ((nf * 4 + 255) / 256) * 256;
    return p;
  };
  auto allocb = [&](size_t nb) -> __bf16* {
    __bf16* p = (__bf16*)(ws + off);
    off += ((nb * 2 + 255) / 256) * 256;
    return p;
  };
  const int M0 = BB * L0;
  // f32 buffers
  float* X     = allocf((size_t)M0 * DM);
  float* Y1    = allocf((size_t)M0 * DM);
  float* Qb    = allocf((size_t)M0 * DM);
  float* Kb    = allocf((size_t)M0 * DM);
  float* Vb    = allocf((size_t)M0 * DM);
  float* CTX   = allocf((size_t)M0 * DM);
  float* F1    = allocf((size_t)M0 * DFF);
  float* MM    = allocf((size_t)BB * NH * L0);
  float* SC    = allocf((size_t)BB * NH * 64 * L0);
  float* MEANV = allocf(BB * DM);
  float* MUV   = allocf(DM);
  float* VARV  = allocf(DM);
  int*   TOP   = (int*)allocf(BB * NH * 64);
  // bf16 weight buffers (converted once per call)
  __bf16* WqB  = allocb((size_t)3 * DM * DM);
  __bf16* WkB  = allocb((size_t)3 * DM * DM);
  __bf16* WvB  = allocb((size_t)3 * DM * DM);
  __bf16* WoB  = allocb((size_t)3 * DM * DM);
  __bf16* C1B_ = allocb((size_t)3 * DFF * DM);
  __bf16* C2B_ = allocb((size_t)3 * DM * DFF);
  __bf16* DCB_ = allocb((size_t)2 * DM * DM * 3);
  // bf16 activation buffers
  __bf16* XB   = allocb((size_t)M0 * DM);
  __bf16* CTXB = allocb((size_t)M0 * DM);
  __bf16* F1B  = allocb((size_t)M0 * DFF);
  __bf16* XCB  = allocb((size_t)M0 * 1536);

  // weight conversions
  cvt(Wq,  WqB,  (size_t)3 * DM * DM, stream);
  cvt(Wk,  WkB,  (size_t)3 * DM * DM, stream);
  cvt(Wv,  WvB,  (size_t)3 * DM * DM, stream);
  cvt(Wo,  WoB,  (size_t)3 * DM * DM, stream);
  cvt(c1w, C1B_, (size_t)3 * DFF * DM, stream);
  cvt(c2w, C2B_, (size_t)3 * DM * DFF, stream);
  cvt(dcw, DCB_, (size_t)2 * DM * DM * 3, stream);

  // embedding
  {
    int tot = M0 * DM;
    k_embed<<<(tot + 255) / 256, 256, 0, stream>>>(x_enc, tok_w, X, L0);
  }

  int Lcur = L0;
  for (int l = 0; l < 3; ++l) {
    const int M = BB * Lcur;
    int U = 5 * (int)ceil(log((double)Lcur));
    if (U > Lcur) U = Lcur;

    dim3 gb(DM / 64, M / 128);
    // Q, K, V projections (WMMA, bf16)
    cvt(X, XB, (size_t)M * DM, stream);
    k_gemm<<<gb, 256, 0, stream>>>(XB, WqB + (size_t)l * DM * DM, bq + l * DM, nullptr, Qb, M, DM, DM, 0);
    k_gemm<<<gb, 256, 0, stream>>>(XB, WkB + (size_t)l * DM * DM, bk + l * DM, nullptr, Kb, M, DM, DM, 0);
    k_gemm<<<gb, 256, 0, stream>>>(XB, WvB + (size_t)l * DM * DM, bv + l * DM, nullptr, Vb, M, DM, DM, 0);

    // ProbSparse attention
    k_probe_scores<<<(BB * NH * Lcur + 255) / 256, 256, 0, stream>>>(Qb, Kb, MM, Lcur, U, l);
    k_topu<<<BB * NH, 256, 0, stream>>>(MM, TOP, Lcur, U);
    k_attn_scores<<<(BB * NH * U * Lcur + 255) / 256, 256, 0, stream>>>(Qb, Kb, TOP, SC, Lcur, U);
    k_softmax<<<BB * NH * U, 256, 0, stream>>>(SC, Lcur);
    k_meanv<<<(BB * DM + 255) / 256, 256, 0, stream>>>(Vb, MEANV, Lcur);
    k_fill_ctx<<<(M * DM + 255) / 256, 256, 0, stream>>>(MEANV, CTX, Lcur);
    k_ctx_update<<<(BB * NH * U * DH + 255) / 256, 256, 0, stream>>>(SC, Vb, TOP, CTX, Lcur, U);

    // output projection + residual, then LN1
    cvt(CTX, CTXB, (size_t)M * DM, stream);
    k_gemm<<<gb, 256, 0, stream>>>(CTXB, WoB + (size_t)l * DM * DM, bo + l * DM, X, Y1, M, DM, DM, 0);
    k_layernorm<<<M, 256, 0, stream>>>(Y1, n1g + l * DM, n1b + l * DM, X);

    // FFN (WMMA, bf16): relu(X@c1^T+b1) @ c2^T + b2 + X, then LN2
    cvt(X, XB, (size_t)M * DM, stream);
    dim3 g1(DFF / 64, M / 128);
    k_gemm<<<g1, 256, 0, stream>>>(XB, C1B_ + (size_t)l * DFF * DM, c1b + l * DFF, nullptr, F1, M, DFF, DM, 1);
    cvt(F1, F1B, (size_t)M * DFF, stream);
    k_gemm<<<gb, 256, 0, stream>>>(F1B, C2B_ + (size_t)l * DM * DFF, c2b + l * DM, X, Y1, M, DM, DFF, 0);
    k_layernorm<<<M, 256, 0, stream>>>(Y1, n2g + l * DM, n2b + l * DM, X);

    // distil conv (WMMA GEMM over gathered taps) + BN + ELU + pool
    if (l < 2) {
      k_gather3<<<(M * 1536 + 255) / 256, 256, 0, stream>>>(X, XCB, Lcur);
      k_gemm<<<gb, 256, 0, stream>>>(XCB, DCB_ + (size_t)l * DM * DM * 3, dcb + l * DM, nullptr, Y1, M, DM, 1536, 0);
      k_bn_stats<<<DM, 256, 0, stream>>>(Y1, MUV, VARV, M);
      int Lo = Lcur / 2;
      k_bn_elu_pool<<<(BB * Lo * DM + 255) / 256, 256, 0, stream>>>(
          Y1, MUV, VARV, bng + l * DM, bnb + l * DM, X, Lcur);
      Lcur = Lo;
    }
  }

  // final layernorm + mean over sequence
  const int Mf = BB * Lcur;
  k_layernorm<<<Mf, 256, 0, stream>>>(X, fng, fnb, Y1);
  k_mean<<<(BB * DM + 255) / 256, 256, 0, stream>>>(Y1, out, Lcur);
}
